// TDGSAttention_2740189135231
// MI455X (gfx1250) — matrix operand order repaired and checked by reading, not compiled, not run on hardware
//
#include <hip/hip_runtime.h>
#include <hip/hip_bf16.h>

#define USE_ASYNC 1

typedef __attribute__((ext_vector_type(16))) _Float16 v16h;
typedef __attribute__((ext_vector_type(8)))  float    v8f;

#define HIDDEN 1024
#define NH  16
#define NHK 4
#define GQ  4
#define HD  64
#define BB  2
#define SKV 4096
#define SQ  512

union AFrag {
    v16h v;
    unsigned int u[8];
    _Float16 h[16];
};

// ---------------------------------------------------------------------------
// f32 -> f16 cast (grid-stride)
// ---------------------------------------------------------------------------
__global__ void cast_f32_f16(const float* __restrict__ in, _Float16* __restrict__ out, int n)
{
    int i = blockIdx.x * blockDim.x + threadIdx.x;
    int stride = gridDim.x * blockDim.x;
    for (; i < n; i += stride) out[i] = (_Float16)in[i];
}

// ---------------------------------------------------------------------------
// C[M,N] = A[M,K] @ W[N,K]^T   (A,W f16 row-major; acc f32)
// Optional fused per-64-col RMS-norm (normw != null), output f16 or f32.
// Block = 128 threads (4 waves). Block tile = 16 x 64; wave tile = 16 x 16.
// Two independent accumulators break the WMMA C-chain.
// ---------------------------------------------------------------------------
__global__ __launch_bounds__(128)
void gemm_wT(const _Float16* __restrict__ A, const _Float16* __restrict__ W,
             int M, int N, int K,
             const float* __restrict__ normw,
             _Float16* __restrict__ outh, float* __restrict__ outf)
{
    __shared__ float Ds[16][64];
    __shared__ float ssq[16];

    const int tid  = threadIdx.x;
    const int wave = tid >> 5;
    const int lane = tid & 31;

    const int tiles_n = N >> 6;
    const int bm = (blockIdx.x / tiles_n) << 4;
    const int bn = (blockIdx.x % tiles_n) << 6;
    const int cb = bn + (wave << 4);          // this wave's 16 output columns

    const int am = lane & 15;
    const int ak = (lane >> 4) << 4;          // A operand: +16 K for hi lane group
    const int bn16 = lane & 15;
    const int bk   = (lane >> 4) << 4;        // B operand: +16 K for hi lane group

    const _Float16* Arow = A + (size_t)(bm + am)   * K;
    const _Float16* Wrow = W + (size_t)(cb + bn16) * K;

    v8f acc0 = {}, acc1 = {};
    for (int k0 = 0; k0 < K; k0 += 64) {
        AFrag a0, b0, a1, b1;
#pragma unroll
        for (int p = 0; p < 8; ++p) {
            int ka = 2 * p + ak;
            a0.u[p] = *(const unsigned int*)(Arow + k0 + ka);
            b0.u[p] = *(const unsigned int*)(Wrow + k0 + 2 * p + bk);
            a1.u[p] = *(const unsigned int*)(Arow + k0 + 32 + ka);
            b1.u[p] = *(const unsigned int*)(Wrow + k0 + 32 + 2 * p + bk);
        }
        acc0 = __builtin_amdgcn_wmma_f32_16x16x32_f16(false, a0.v, false, b0.v,
                                                      (short)0, acc0, false, false);
        acc1 = __builtin_amdgcn_wmma_f32_16x16x32_f16(false, a1.v, false, b1.v,
                                                      (short)0, acc1, false, false);
    }
    v8f acc = acc0 + acc1;

    // C/D layout: lane col n = lane&15, rows r + (lane>=16 ? 8 : 0)
    const int ro  = (lane >> 4) << 3;
    const int col = (wave << 4) + (lane & 15);
#pragma unroll
    for (int r = 0; r < 8; ++r) Ds[r + ro][col] = acc[r];
    __syncthreads();

    if (normw && tid < 16) {
        float s = 0.f;
#pragma unroll
        for (int c = 0; c < 64; ++c) { float x = Ds[tid][c]; s += x * x; }
        ssq[tid] = s;
    }
    __syncthreads();

#pragma unroll
    for (int i = 0; i < 8; ++i) {
        int flat = tid * 8 + i;
        int m = flat >> 6, c = flat & 63;
        float v = Ds[m][c];
        if (normw) v = v * rsqrtf(ssq[m] * (1.0f / 64.0f) + 1e-6f) * normw[c];
        size_t o = (size_t)(bm + m) * N + bn + c;
        if (outh) outh[o] = (_Float16)v;
        else      outf[o] = v;
    }
}

// ---------------------------------------------------------------------------
// Flash attention: block = 4 waves sharing one K/V chunk; each wave owns a
// 16-row q tile of the same head. KV streamed in 32-row chunks via async
// loads to LDS (ASYNCcnt). Q operands built directly from global.
// Q: [B*SQ][NH*HD] f16 (rms-normed), K/V: [B*SKV][NHK*HD] f16, O: f16.
// ---------------------------------------------------------------------------
__global__ __launch_bounds__(128)
void attn_kernel(const _Float16* __restrict__ Q, const _Float16* __restrict__ Kx,
                 const _Float16* __restrict__ V, _Float16* __restrict__ O)
{
    __shared__ _Float16 Ks[32][64];
    __shared__ _Float16 Vs[32][64];
    __shared__ _Float16 Ps[4][16][32];

    const int tid  = threadIdx.x;
    const int wave = tid >> 5;
    const int lane = tid & 31;

    const int qblocks = SQ / 64;              // 8 blocks of 64 q-rows per (b,h)
    const int qb = blockIdx.x % qblocks;
    const int h  = (blockIdx.x / qblocks) % NH;
    const int b  = blockIdx.x / (qblocks * NH);
    const int hk = h / GQ;
    const int q0 = qb * 64 + wave * 16;       // this wave's 16 q rows

    // ---- build Q A-operands directly from global (pairs contiguous) ----
    AFrag aq0, aq1;
    {
        const _Float16* Qrow =
            Q + (size_t)(b * SQ + q0 + (lane & 15)) * (NH * HD) + h * HD;
        int ko = (lane >> 4) << 4;
#pragma unroll
        for (int p = 0; p < 8; ++p) {
            int kk = 2 * p + ko;
            aq0.u[p] = *(const unsigned int*)(Qrow + kk);
            aq1.u[p] = *(const unsigned int*)(Qrow + kk + 32);
        }
    }

    float mrow[8], lrow[8];
    v8f o0 = {}, o1 = {}, o2 = {}, o3 = {};
#pragma unroll
    for (int r = 0; r < 8; ++r) { mrow[r] = -3.0e38f; lrow[r] = 0.f; }

    const _Float16* Kb = Kx + (size_t)(b * SKV) * (NHK * HD) + hk * HD;
    const _Float16* Vb = V  + (size_t)(b * SKV) * (NHK * HD) + hk * HD;

    // cooperative K/V chunk load: 128 threads x 16 halves per array
    const int flat0 = tid * 16;
    const int lr = flat0 >> 6;                // chunk row 0..31
    const int ld = flat0 & 63;                // col 0,16,32,48

    for (int s0 = 0; s0 < SKV; s0 += 32) {
        const _Float16* kr = Kb + (size_t)(s0 + lr) * (NHK * HD) + ld;
        const _Float16* vr = Vb + (size_t)(s0 + lr) * (NHK * HD) + ld;
#if USE_ASYNC
        {
            unsigned kdst = (unsigned)(unsigned long long)&Ks[lr][ld];
            unsigned vdst = (unsigned)(unsigned long long)&Vs[lr][ld];
            unsigned long long kga = (unsigned long long)kr;
            unsigned long long vga = (unsigned long long)vr;
            asm volatile(
                "global_load_async_to_lds_b128 %0, %2, off\n\t"
                "global_load_async_to_lds_b128 %0, %2, off offset:16\n\t"
                "global_load_async_to_lds_b128 %1, %3, off\n\t"
                "global_load_async_to_lds_b128 %1, %3, off offset:16"
                :: "v"(kdst), "v"(vdst), "v"(kga), "v"(vga)
                : "memory");
        }
#else
        *(uint4*)&Ks[lr][ld]     = *(const uint4*)(kr);
        *(uint4*)&Ks[lr][ld + 8] = *(const uint4*)(kr + 8);
        *(uint4*)&Vs[lr][ld]     = *(const uint4*)(vr);
        *(uint4*)&Vs[lr][ld + 8] = *(const uint4*)(vr + 8);
#endif
        if (s0 + 32 < SKV) {
            __builtin_prefetch(kr + (size_t)32 * (NHK * HD), 0, 1);
            __builtin_prefetch(vr + (size_t)32 * (NHK * HD), 0, 1);
        }
#if USE_ASYNC
        asm volatile("s_wait_asynccnt 0x0" ::: "memory");
#endif
        __syncthreads();

        // ---- S = Q @ K^T for 32 kv columns: two 16x16 C tiles ----
        v8f s_t0 = {}, s_t1 = {};
        {
            int n  = lane & 15;
            int ko = (lane >> 4) << 4;
            AFrag b0, b1;
#pragma unroll
            for (int p = 0; p < 8; ++p) {
                b0.u[p] = *(const unsigned int*)&Ks[n][2 * p + ko];
                b1.u[p] = *(const unsigned int*)&Ks[n][2 * p + ko + 32];
            }
            s_t0 = __builtin_amdgcn_wmma_f32_16x16x32_f16(false, aq0.v, false, b0.v,
                                                          (short)0, s_t0, false, false);
            s_t0 = __builtin_amdgcn_wmma_f32_16x16x32_f16(false, aq1.v, false, b1.v,
                                                          (short)0, s_t0, false, false);
#pragma unroll
            for (int p = 0; p < 8; ++p) {
                b0.u[p] = *(const unsigned int*)&Ks[16 + n][2 * p + ko];
                b1.u[p] = *(const unsigned int*)&Ks[16 + n][2 * p + ko + 32];
            }
            s_t1 = __builtin_amdgcn_wmma_f32_16x16x32_f16(false, aq0.v, false, b0.v,
                                                          (short)0, s_t1, false, false);
            s_t1 = __builtin_amdgcn_wmma_f32_16x16x32_f16(false, aq1.v, false, b1.v,
                                                          (short)0, s_t1, false, false);
        }

        // ---- online softmax (rows live across 16-lane groups) ----
        {
            int ro = (lane >> 4) << 3;
            int nc = lane & 15;
#pragma unroll
            for (int r = 0; r < 8; ++r) {
                float t = fmaxf(s_t0[r], s_t1[r]);
                t = fmaxf(t, __shfl_xor(t, 1));
                t = fmaxf(t, __shfl_xor(t, 2));
                t = fmaxf(t, __shfl_xor(t, 4));
                t = fmaxf(t, __shfl_xor(t, 8));
                float mn = fmaxf(mrow[r], t);
                float sc = __expf(mrow[r] - mn);
                float e0 = __expf(s_t0[r] - mn);
                float e1 = __expf(s_t1[r] - mn);
                float rs = e0 + e1;
                rs += __shfl_xor(rs, 1);
                rs += __shfl_xor(rs, 2);
                rs += __shfl_xor(rs, 4);
                rs += __shfl_xor(rs, 8);
                lrow[r] = lrow[r] * sc + rs;
                mrow[r] = mn;
                o0[r] *= sc; o1[r] *= sc; o2[r] *= sc; o3[r] *= sc;
                Ps[wave][r + ro][nc]      = (_Float16)e0;
                Ps[wave][r + ro][nc + 16] = (_Float16)e1;
            }
        }
        __syncthreads();

        // ---- O += P @ V  (16x32 @ 32x64 -> four 16x16 C tiles) ----
        {
            AFrag ap;
            int m  = lane & 15;
            int ko = (lane >> 4) << 4;
#pragma unroll
            for (int p = 0; p < 8; ++p) {
                int kk = 2 * p + ko;
                ap.u[p] = *(const unsigned int*)&Ps[wave][m][kk];
            }
            int n   = lane & 15;
            int kvo = (lane >> 4) << 4;
            AFrag bv;
#pragma unroll
            for (int hh = 0; hh < 16; ++hh) bv.h[hh] = Vs[hh + kvo][n];
            o0 = __builtin_amdgcn_wmma_f32_16x16x32_f16(false, ap.v, false, bv.v,
                                                        (short)0, o0, false, false);
#pragma unroll
            for (int hh = 0; hh < 16; ++hh) bv.h[hh] = Vs[hh + kvo][16 + n];
            o1 = __builtin_amdgcn_wmma_f32_16x16x32_f16(false, ap.v, false, bv.v,
                                                        (short)0, o1, false, false);
#pragma unroll
            for (int hh = 0; hh < 16; ++hh) bv.h[hh] = Vs[hh + kvo][32 + n];
            o2 = __builtin_amdgcn_wmma_f32_16x16x32_f16(false, ap.v, false, bv.v,
                                                        (short)0, o2, false, false);
#pragma unroll
            for (int hh = 0; hh < 16; ++hh) bv.h[hh] = Vs[hh + kvo][48 + n];
            o3 = __builtin_amdgcn_wmma_f32_16x16x32_f16(false, ap.v, false, bv.v,
                                                        (short)0, o3, false, false);
        }
        __syncthreads();
    }

    // ---- epilogue: O /= l, write f16 ----
    _Float16* Ob = O + (size_t)(b * SQ + q0) * (NH * HD) + h * HD;
    {
        int ro = (lane >> 4) << 3;
        int n  = lane & 15;
#pragma unroll
        for (int r = 0; r < 8; ++r) {
            float ri = 1.0f / lrow[r];
            size_t base = (size_t)(r + ro) * (NH * HD);
            Ob[base + n]      = (_Float16)(o0[r] * ri);
            Ob[base + 16 + n] = (_Float16)(o1[r] * ri);
            Ob[base + 32 + n] = (_Float16)(o2[r] * ri);
            Ob[base + 48 + n] = (_Float16)(o3[r] * ri);
        }
    }
}

// ---------------------------------------------------------------------------
extern "C" void kernel_launch(void* const* d_in, const int* in_sizes, int n_in,
                              void* d_out, int out_size, void* d_ws, size_t ws_size,
                              hipStream_t stream)
{
    (void)in_sizes; (void)n_in; (void)out_size; (void)ws_size;

    const float* inputs = (const float*)d_in[0];
    const float* latent = (const float*)d_in[1];
    const float* wq = (const float*)d_in[2];
    const float* wk = (const float*)d_in[3];
    const float* wv = (const float*)d_in[4];
    const float* wo = (const float*)d_in[5];
    const float* qnw = (const float*)d_in[6];
    const float* knw = (const float*)d_in[7];

    _Float16* ws = (_Float16*)d_ws;
    size_t off = 0;
    _Float16* inputs_h = ws + off; off += (size_t)BB * SKV * HIDDEN;
    _Float16* latent_h = ws + off; off += (size_t)BB * SQ * HIDDEN;
    _Float16* wq_h = ws + off; off += (size_t)NH  * HD * HIDDEN;
    _Float16* wk_h = ws + off; off += (size_t)NHK * HD * HIDDEN;
    _Float16* wv_h = ws + off; off += (size_t)NHK * HD * HIDDEN;
    _Float16* wo_h = ws + off; off += (size_t)HIDDEN * NH * HD;
    _Float16* Qn = ws + off; off += (size_t)BB * SQ  * NH  * HD;
    _Float16* Kn = ws + off; off += (size_t)BB * SKV * NHK * HD;
    _Float16* Vh = ws + off; off += (size_t)BB * SKV * NHK * HD;
    _Float16* AO = ws + off; off += (size_t)BB * SQ  * NH  * HD;

    auto cast = [&](const float* src, _Float16* dst, int n) {
        int blocks = (n + 255) / 256;
        if (blocks > 4096) blocks = 4096;
        cast_f32_f16<<<blocks, 256, 0, stream>>>(src, dst, n);
    };
    cast(inputs, inputs_h, BB * SKV * HIDDEN);
    cast(latent, latent_h, BB * SQ * HIDDEN);
    cast(wq, wq_h, NH  * HD * HIDDEN);
    cast(wk, wk_h, NHK * HD * HIDDEN);
    cast(wv, wv_h, NHK * HD * HIDDEN);
    cast(wo, wo_h, HIDDEN * NH * HD);

    // Q projection + rmsnorm -> Qn
    gemm_wT<<<(BB * SQ / 16) * (NH * HD / 64), 128, 0, stream>>>(
        latent_h, wq_h, BB * SQ, NH * HD, HIDDEN, qnw, Qn, nullptr);
    // K projection + rmsnorm -> Kn
    gemm_wT<<<(BB * SKV / 16) * (NHK * HD / 64), 128, 0, stream>>>(
        inputs_h, wk_h, BB * SKV, NHK * HD, HIDDEN, knw, Kn, nullptr);
    // V projection -> Vh
    gemm_wT<<<(BB * SKV / 16) * (NHK * HD / 64), 128, 0, stream>>>(
        inputs_h, wv_h, BB * SKV, NHK * HD, HIDDEN, nullptr, Vh, nullptr);
    // attention -> AO
    attn_kernel<<<BB * NH * (SQ / 64), 128, 0, stream>>>(Qn, Kn, Vh, AO);
    // output projection -> d_out (f32)
    gemm_wT<<<(BB * SQ / 16) * (HIDDEN / 64), 128, 0, stream>>>(
        AO, wo_h, BB * SQ, HIDDEN, NH * HD, nullptr, nullptr, (float*)d_out);
}
